// GraphTransformer_48704929137094
// MI455X (gfx1250) — compile-verified, matrix operand order checked by reading, not run on hardware
//
#include <hip/hip_runtime.h>
#include <math.h>

#define NN_   60000
#define EE_   300000
#define GG_   120
#define IN_   128
#define HH_   128
#define NHEAD 8
#define DHEAD 16
#define LL_   4
#define MLP_  512
#define OUT_  6

typedef __attribute__((ext_vector_type(16))) _Float16 v16h;
typedef __attribute__((ext_vector_type(8)))  _Float16 v8h;
typedef __attribute__((ext_vector_type(4)))  _Float16 v4h;
typedef __attribute__((ext_vector_type(8)))  float    v8f;

// ---------------- utility device funcs ----------------

__device__ __forceinline__ unsigned f2key(float x) {
    unsigned vu = __float_as_uint(x);
    return (vu & 0x80000000u) ? ~vu : (vu | 0x80000000u);
}
__device__ __forceinline__ float key2f(unsigned k) {
    unsigned vu = (k & 0x80000000u) ? (k & 0x7FFFFFFFu) : ~k;
    return __uint_as_float(vu);
}
__device__ __forceinline__ float wave_sum32(float v) {
#pragma unroll
    for (int off = 16; off > 0; off >>= 1) v += __shfl_xor(v, off, 32);
    return v;
}
__device__ __forceinline__ float gelu_exact(float x) {
    return 0.5f * x * (1.0f + erff(x * 0.70710678118654752f));
}

// ---------------- fill / convert ----------------

__global__ void fill_u32_k(unsigned* __restrict__ p, unsigned v, long n) {
    long i = (long)blockIdx.x * blockDim.x + threadIdx.x;
    if (i < n) p[i] = v;
}

// f32 -> f16, 4 elements per thread (b128 in, b64 out)
__global__ void cvt_f16_k(const float* __restrict__ in, _Float16* __restrict__ out, long n4) {
    long i = (long)blockIdx.x * blockDim.x + threadIdx.x;
    if (i >= n4) return;
    float4 v = ((const float4*)in)[i];
    ((v4h*)out)[i] = (v4h){(_Float16)v.x, (_Float16)v.y, (_Float16)v.z, (_Float16)v.w};
}

// ---------------- segment-min of node index per graph ----------------

__global__ void seg_first_k(const int* __restrict__ batch, int* __restrict__ first, int n) {
    int i = blockIdx.x * blockDim.x + threadIdx.x;
    if (i < n) atomicMin(&first[batch[i]], i);
}

// ---------------- weight pre-pack into WMMA B-fragment layout ----------------
// Fragment (col tile ct, k-step ks): lane l, half t ->
//   K = ks*32 + ((l>>4)<<3) + (t<8 ? t : t+8),  N = ct*16 + (l&15)
// Packed index: (((ct*(K_/32) + ks)*32 + l)*16 + t)

template <int K_, int NOUT>
__global__ void pack_b_k(const float* __restrict__ B, _Float16* __restrict__ P) {
    int i = blockIdx.x * blockDim.x + threadIdx.x;
    if (i >= K_ * NOUT) return;
    const int t    = i & 15;
    const int lane = (i >> 4) & 31;
    const int frag = i >> 9;
    const int ks   = frag % (K_ / 32);
    const int ct   = frag / (K_ / 32);
    const int k = ks * 32 + ((lane >> 4) << 3) + (t < 8 ? t : t + 8);
    const int n = ct * 16 + (lane & 15);
    P[i] = (_Float16)B[(size_t)k * NOUT + n];
}

// ---------------- WMMA GEMM ----------------
// C = A16[m,K_] @ Bpacked + bias, epilogue. A16 row-major f16; B pre-packed.
// EPI: 0 none, 1 exact GELU, 2 + pos_emb[idx - first[batch[idx]]]
// WF32/WF16 select f32 / f16 outputs.

template <int EPI, int K_, int NOUT, int MT, bool WF32, bool WF16>
__global__ __launch_bounds__(256) void gemm_wmma_k(
    const _Float16* __restrict__ A16, const _Float16* __restrict__ BP,
    const float* __restrict__ bias, float* __restrict__ C,
    _Float16* __restrict__ C16,
    const float* __restrict__ pos_emb, const int* __restrict__ batch,
    const int* __restrict__ first)
{
    const int lane = threadIdx.x & 31;
    const int wave = threadIdx.x >> 5;
    const int row0 = blockIdx.x * (16 * MT);
    const int colt = blockIdx.y * 8 + wave;
    const int col0 = colt * 16;

    v8f acc[MT];
#pragma unroll
    for (int mt = 0; mt < MT; ++mt) acc[mt] = (v8f){};

    const int mloc = lane & 15;
    const int nn   = col0 + mloc;
    const int koff = (lane >> 4) << 3;  // 0 or 8

    const _Float16* bfrag = BP + ((size_t)colt * (K_ / 32)) * 512 + lane * 16;

#pragma unroll 4
    for (int ks = 0; ks < K_ / 32; ++ks) {
        const v16h b = *(const v16h*)(bfrag + (size_t)ks * 512);   // 32B aligned
        const int kb = ks * 32 + koff;
#pragma unroll
        for (int mt = 0; mt < MT; ++mt) {
            const _Float16* ap = A16 + (size_t)(row0 + mt * 16 + mloc) * K_ + kb;
            const v8h lo = *(const v8h*)ap;          // K = kb .. kb+7
            const v8h hi = *(const v8h*)(ap + 16);   // K = kb+16 .. kb+23
            const v16h a = __builtin_shufflevector(lo, hi,
                0, 1, 2, 3, 4, 5, 6, 7, 8, 9, 10, 11, 12, 13, 14, 15);
            acc[mt] = __builtin_amdgcn_wmma_f32_16x16x32_f16(
                false, a, false, b, (short)0, acc[mt], false, false);
        }
    }

    // C/D layout: acc[mt][r] -> M = row0 + mt*16 + r + (lane>=16 ? 8 : 0), N = nn
    const float bvl = bias[nn];
#pragma unroll
    for (int mt = 0; mt < MT; ++mt) {
        const int mbase = row0 + mt * 16 + ((lane >> 4) << 3);
#pragma unroll
        for (int r = 0; r < 8; ++r) {
            const int mm = mbase + r;
            float v = acc[mt][r] + bvl;
            if (EPI == 1) {
                v = gelu_exact(v);
            } else if (EPI == 2) {
                const int pos = mm - first[batch[mm]];
                v += pos_emb[(size_t)pos * HH_ + nn];
            }
            if (WF32) C[(size_t)mm * NOUT + nn] = v;
            if (WF16) C16[(size_t)mm * NOUT + nn] = (_Float16)v;
        }
    }
}

// ---------------- edge attention ----------------

__global__ void edge_scores_k(const float* __restrict__ Q, const float* __restrict__ Kv,
                              const int* __restrict__ row, const int* __restrict__ col,
                              float* __restrict__ scores, unsigned* __restrict__ mkey) {
    int e = blockIdx.x * blockDim.x + threadIdx.x;
    if (e >= EE_) return;
    const int r = row[e], c = col[e];
    const float* q = Q + (size_t)r * HH_;
    const float* k = Kv + (size_t)c * HH_;
    float mx = -3.4e38f;
#pragma unroll
    for (int h = 0; h < NHEAD; ++h) {
        float d = 0.f;
#pragma unroll
        for (int t = 0; t < DHEAD; ++t) d += q[h * DHEAD + t] * k[h * DHEAD + t];
        d *= 0.25f;  // 1/sqrt(16)
        scores[(size_t)e * NHEAD + h] = d;
        mx = fmaxf(mx, d);
    }
    atomicMax(&mkey[r], f2key(mx));
}

__global__ void edge_exp_k(const int* __restrict__ row,
                           const unsigned* __restrict__ mkey,
                           float* __restrict__ ex, float* __restrict__ s) {
    long i = (long)blockIdx.x * blockDim.x + threadIdx.x;
    if (i >= (long)EE_ * NHEAD) return;
    const int e = (int)(i / NHEAD);
    const int h = (int)(i % NHEAD);
    const int r = row[e];
    const float m = key2f(mkey[r]);
    const float v = expf(ex[i] - m);
    ex[i] = v;
    atomicAdd(&s[(size_t)r * NHEAD + h], v);
}

__global__ void edge_agg_k(const int* __restrict__ row, const int* __restrict__ col,
                           const float* __restrict__ ex, const float* __restrict__ s,
                           const float* __restrict__ V, float* __restrict__ agg) {
    long i = (long)blockIdx.x * blockDim.x + threadIdx.x;
    if (i >= (long)EE_ * NHEAD) return;
    const int e = (int)(i / NHEAD);
    const int h = (int)(i % NHEAD);
    const int r = row[e], c = col[e];
    const float w = ex[i] / (s[(size_t)r * NHEAD + h] + 1e-8f);
    const float* vp = V + (size_t)c * HH_ + h * DHEAD;
    float* ap = agg + (size_t)r * HH_ + h * DHEAD;
#pragma unroll
    for (int t = 0; t < DHEAD; ++t) atomicAdd(&ap[t], w * vp[t]);
}

// ---------------- residual + layernorm ----------------
// one row per wave32; lane owns 4 contiguous elements (float4 loads);
// writes f32 into h and f16 copy into h16; add == nullptr -> plain LN

__global__ __launch_bounds__(256) void resid_ln_k(float* __restrict__ h,
                                                  _Float16* __restrict__ h16,
                                                  const float* __restrict__ add,
                                                  const float* __restrict__ sc,
                                                  const float* __restrict__ bi, int n) {
    const int lane = threadIdx.x & 31;
    const int wave = threadIdx.x >> 5;
    const int r = blockIdx.x * 8 + wave;
    if (r >= n) return;
    const int base = lane * 4;
    float4 v = *(const float4*)(h + (size_t)r * HH_ + base);
    if (add) {
        float4 a = *(const float4*)(add + (size_t)r * HH_ + base);
        v.x += a.x; v.y += a.y; v.z += a.z; v.w += a.w;
    }
    const float sum = wave_sum32(v.x + v.y + v.z + v.w);
    const float mu = sum * (1.0f / HH_);
    float vs = (v.x - mu) * (v.x - mu) + (v.y - mu) * (v.y - mu) +
               (v.z - mu) * (v.z - mu) + (v.w - mu) * (v.w - mu);
    vs = wave_sum32(vs);
    const float rstd = rsqrtf(vs * (1.0f / HH_) + 1e-5f);
    const float4 s4 = *(const float4*)(sc + base);
    const float4 b4 = *(const float4*)(bi + base);
    float4 o;
    o.x = (v.x - mu) * rstd * s4.x + b4.x;
    o.y = (v.y - mu) * rstd * s4.y + b4.y;
    o.z = (v.z - mu) * rstd * s4.z + b4.z;
    o.w = (v.w - mu) * rstd * s4.w + b4.w;
    *(float4*)(h + (size_t)r * HH_ + base) = o;
    *(v4h*)(h16 + (size_t)r * HH_ + base) =
        (v4h){(_Float16)o.x, (_Float16)o.y, (_Float16)o.z, (_Float16)o.w};
}

// ---------------- pooling ----------------

__global__ void pool_k(const float* __restrict__ h, const int* __restrict__ batch,
                       float* __restrict__ pooled, float* __restrict__ cnt) {
    const int i = blockIdx.x;
    const int j = threadIdx.x;
    const int b = batch[i];
    atomicAdd(&pooled[(size_t)b * HH_ + j], h[(size_t)i * HH_ + j]);
    if (j == 0) atomicAdd(&cnt[b], 1.0f);
}

// ---------------- prediction head (tiny) ----------------

__global__ void head_k(const float* __restrict__ pooled, const float* __restrict__ cnt,
                       const float* __restrict__ Wp1, const float* __restrict__ bp1,
                       const float* __restrict__ Wp2, const float* __restrict__ bp2,
                       float* __restrict__ out) {
    __shared__ float prow[HH_];
    __shared__ float p1[HH_ / 2];
    const int g = blockIdx.x;
    const int t = threadIdx.x;   // 128 threads
    const float ic = 1.0f / fmaxf(cnt[g], 1.0f);
    prow[t] = pooled[(size_t)g * HH_ + t] * ic;
    __syncthreads();
    if (t < HH_ / 2) {
        float acc = bp1[t];
#pragma unroll 4
        for (int k = 0; k < HH_; ++k) acc += prow[k] * Wp1[k * (HH_ / 2) + t];
        p1[t] = fmaxf(acc, 0.f);
    }
    __syncthreads();
    if (t < OUT_) {
        float o = bp2[t];
#pragma unroll 4
        for (int k = 0; k < HH_ / 2; ++k) o += p1[k] * Wp2[k * OUT_ + t];
        out[(size_t)g * OUT_ + t] = o;
    }
}

// ---------------- launch ----------------

extern "C" void kernel_launch(void* const* d_in, const int* in_sizes, int n_in,
                              void* d_out, int out_size, void* d_ws, size_t ws_size,
                              hipStream_t stream) {
    const float* x       = (const float*)d_in[0];
    const int*   eidx    = (const int*)d_in[1];
    const int*   batch   = (const int*)d_in[2];
    const float* Win     = (const float*)d_in[3];
    const float* bin_    = (const float*)d_in[4];
    const float* pos_emb = (const float*)d_in[5];
    const float* Wq      = (const float*)d_in[6];
    const float* bq      = (const float*)d_in[7];
    const float* Wk      = (const float*)d_in[8];
    const float* bk      = (const float*)d_in[9];
    const float* Wv      = (const float*)d_in[10];
    const float* bv      = (const float*)d_in[11];
    const float* Wo      = (const float*)d_in[12];
    const float* bo      = (const float*)d_in[13];
    const float* ln1_s   = (const float*)d_in[14];
    const float* ln1_b   = (const float*)d_in[15];
    const float* W1      = (const float*)d_in[16];
    const float* b1      = (const float*)d_in[17];
    const float* W2      = (const float*)d_in[18];
    const float* b2      = (const float*)d_in[19];
    const float* ln2_s   = (const float*)d_in[20];
    const float* ln2_b   = (const float*)d_in[21];
    const float* fn_s    = (const float*)d_in[22];
    const float* fn_b    = (const float*)d_in[23];
    const float* Wp1     = (const float*)d_in[24];
    const float* bp1     = (const float*)d_in[25];
    const float* Wp2     = (const float*)d_in[26];
    const float* bp2     = (const float*)d_in[27];

    const int* row = eidx;          // edge_index[0]
    const int* col = eidx + EE_;    // edge_index[1]

    // ---- workspace arena ----
    char* ws = (char*)d_ws;
    const size_t SZ  = (size_t)NN_ * HH_ * sizeof(float);     // 30.72 MB
    const size_t SZH = (size_t)NN_ * HH_ * sizeof(_Float16);  // 15.36 MB
    float*     h_     = (float*)(ws);
    float*     Qb     = (float*)(ws + 1 * SZ);
    float*     Kb     = (float*)(ws + 2 * SZ);
    float*     Vb     = (float*)(ws + 3 * SZ);
    float*     agg    = (float*)(ws + 4 * SZ);
    float*     aout   = (float*)(ws + 5 * SZ);     // attn-out / m2
    _Float16*  m1h    = (_Float16*)Qb;             // N*MLP f16 == 2*SZ (aliases Q,K; dead then)
    char*      p      = ws + 6 * SZ;
    _Float16*  h16    = (_Float16*)p;  p += SZH;
    _Float16*  xa16   = (_Float16*)p;  p += SZH;   // x16 at start, agg16 per layer
    float*     ex     = (float*)p;     p += (size_t)EE_ * NHEAD * sizeof(float);
    float*     ssum   = (float*)p;     p += (size_t)NN_ * NHEAD * sizeof(float);
    unsigned*  mkey   = (unsigned*)p;  p += (size_t)NN_ * sizeof(unsigned);
    int*       first  = (int*)p;       p += 256;
    float*     pooled = (float*)p;     p += (size_t)GG_ * HH_ * sizeof(float);
    float*     cnt    = (float*)p;     p += 256;
    // packed f16 weights
    _Float16*  WinP   = (_Float16*)p;  p += (size_t)IN_ * HH_ * 2;
    _Float16*  WqP    = (_Float16*)p;  p += (size_t)LL_ * HH_ * HH_ * 2;
    _Float16*  WkP    = (_Float16*)p;  p += (size_t)LL_ * HH_ * HH_ * 2;
    _Float16*  WvP    = (_Float16*)p;  p += (size_t)LL_ * HH_ * HH_ * 2;
    _Float16*  WoP    = (_Float16*)p;  p += (size_t)LL_ * HH_ * HH_ * 2;
    _Float16*  W1P    = (_Float16*)p;  p += (size_t)LL_ * HH_ * MLP_ * 2;
    _Float16*  W2P    = (_Float16*)p;  p += (size_t)LL_ * MLP_ * HH_ * 2;

    const int T = 256;
    const int MT = 2;
    const unsigned GRID_M = NN_ / (16 * MT);  // 1875
    auto blks = [](long n, int t) { return (unsigned)((n + t - 1) / t); };

    // ---- pre-pack all weights to f16 fragment layout (tiny) ----
    pack_b_k<IN_, HH_><<<blks(IN_ * HH_, T), T, 0, stream>>>(Win, WinP);
    for (int l = 0; l < LL_; ++l) {
        pack_b_k<HH_, HH_><<<blks(HH_ * HH_, T), T, 0, stream>>>(
            Wq + (size_t)l * HH_ * HH_, WqP + (size_t)l * HH_ * HH_);
        pack_b_k<HH_, HH_><<<blks(HH_ * HH_, T), T, 0, stream>>>(
            Wk + (size_t)l * HH_ * HH_, WkP + (size_t)l * HH_ * HH_);
        pack_b_k<HH_, HH_><<<blks(HH_ * HH_, T), T, 0, stream>>>(
            Wv + (size_t)l * HH_ * HH_, WvP + (size_t)l * HH_ * HH_);
        pack_b_k<HH_, HH_><<<blks(HH_ * HH_, T), T, 0, stream>>>(
            Wo + (size_t)l * HH_ * HH_, WoP + (size_t)l * HH_ * HH_);
        pack_b_k<HH_, MLP_><<<blks(HH_ * MLP_, T), T, 0, stream>>>(
            W1 + (size_t)l * HH_ * MLP_, W1P + (size_t)l * HH_ * MLP_);
        pack_b_k<MLP_, HH_><<<blks(MLP_ * HH_, T), T, 0, stream>>>(
            W2 + (size_t)l * MLP_ * HH_, W2P + (size_t)l * MLP_ * HH_);
    }

    // first-node-of-graph (segment_min)
    fill_u32_k<<<blks(256, T), T, 0, stream>>>((unsigned*)first, 0x7FFFFFFFu, GG_);
    seg_first_k<<<blks(NN_, T), T, 0, stream>>>(batch, first, NN_);

    // x -> f16, then h = x @ Win + bin + pos_emb[pos]  (writes f32 h and f16 h16)
    cvt_f16_k<<<blks((long)NN_ * IN_ / 4, T), T, 0, stream>>>(x, xa16, (long)NN_ * IN_ / 4);
    gemm_wmma_k<2, IN_, HH_, MT, true, true><<<dim3(GRID_M, 1), T, 0, stream>>>(
        xa16, WinP, bin_, h_, h16, pos_emb, batch, first);

    for (int l = 0; l < LL_; ++l) {
        // Q/K/V projections (f32 outputs for edge kernels)
        gemm_wmma_k<0, HH_, HH_, MT, true, false><<<dim3(GRID_M, 1), T, 0, stream>>>(
            h16, WqP + (size_t)l * HH_ * HH_, bq + l * HH_, Qb, nullptr, nullptr, nullptr, nullptr);
        gemm_wmma_k<0, HH_, HH_, MT, true, false><<<dim3(GRID_M, 1), T, 0, stream>>>(
            h16, WkP + (size_t)l * HH_ * HH_, bk + l * HH_, Kb, nullptr, nullptr, nullptr, nullptr);
        gemm_wmma_k<0, HH_, HH_, MT, true, false><<<dim3(GRID_M, 1), T, 0, stream>>>(
            h16, WvP + (size_t)l * HH_ * HH_, bv + l * HH_, Vb, nullptr, nullptr, nullptr, nullptr);

        // attention over edges
        fill_u32_k<<<blks(NN_, T), T, 0, stream>>>(mkey, 0u, NN_);
        fill_u32_k<<<blks((long)NN_ * NHEAD, T), T, 0, stream>>>((unsigned*)ssum, 0u,
                                                                 (long)NN_ * NHEAD);
        fill_u32_k<<<blks((long)NN_ * HH_, T), T, 0, stream>>>((unsigned*)agg, 0u,
                                                               (long)NN_ * HH_);
        edge_scores_k<<<blks(EE_, T), T, 0, stream>>>(Qb, Kb, row, col, ex, mkey);
        edge_exp_k<<<blks((long)EE_ * NHEAD, T), T, 0, stream>>>(row, mkey, ex, ssum);
        edge_agg_k<<<blks((long)EE_ * NHEAD, T), T, 0, stream>>>(row, col, ex, ssum, Vb, agg);

        // output projection + residual LN
        cvt_f16_k<<<blks((long)NN_ * HH_ / 4, T), T, 0, stream>>>(agg, xa16, (long)NN_ * HH_ / 4);
        gemm_wmma_k<0, HH_, HH_, MT, true, false><<<dim3(GRID_M, 1), T, 0, stream>>>(
            xa16, WoP + (size_t)l * HH_ * HH_, bo + l * HH_, aout, nullptr, nullptr, nullptr, nullptr);
        resid_ln_k<<<blks(NN_, 8), T, 0, stream>>>(h_, h16, aout,
                                                   ln1_s + l * HH_, ln1_b + l * HH_, NN_);

        // MLP: m1h (f16, aliases Q+K) = gelu(h @ W1 + b1); m2 = m1h @ W2 + b2
        gemm_wmma_k<1, HH_, MLP_, MT, false, true><<<dim3(GRID_M, MLP_ / 128), T, 0, stream>>>(
            h16, W1P + (size_t)l * HH_ * MLP_, b1 + l * MLP_, nullptr, m1h,
            nullptr, nullptr, nullptr);
        gemm_wmma_k<0, MLP_, HH_, MT, true, false><<<dim3(GRID_M, 1), T, 0, stream>>>(
            m1h, W2P + (size_t)l * MLP_ * HH_, b2 + l * HH_, aout, nullptr,
            nullptr, nullptr, nullptr);
        resid_ln_k<<<blks(NN_, 8), T, 0, stream>>>(h_, h16, aout,
                                                   ln2_s + l * HH_, ln2_b + l * HH_, NN_);
    }

    // final norm, mean-pool, head
    resid_ln_k<<<blks(NN_, 8), T, 0, stream>>>(h_, h16, nullptr, fn_s, fn_b, NN_);
    fill_u32_k<<<blks((long)GG_ * HH_, T), T, 0, stream>>>((unsigned*)pooled, 0u, (long)GG_ * HH_);
    fill_u32_k<<<blks(256, T), T, 0, stream>>>((unsigned*)cnt, 0u, GG_);
    pool_k<<<NN_, HH_, 0, stream>>>(h_, batch, pooled, cnt);
    head_k<<<GG_, HH_, 0, stream>>>(pooled, cnt, Wp1, bp1, Wp2, bp2, (float*)d_out);

    (void)in_sizes; (void)n_in; (void)out_size; (void)ws_size;
}
